// SparseMOE_30640296689951
// MI455X (gfx1250) — compile-verified, matrix operand order checked by reading, not run on hardware
//
#include <hip/hip_runtime.h>
#include <hip/hip_bf16.h>
#include <math.h>

// ---------------------------------------------------------------------------
// Sparse MoE for MI455X (gfx1250, wave32, WMMA).
// x:[T,H] f32, We:[E,H,H] f32, be:[E,H], Wr:[H,E], br:[E], top_k=2.
// zero -> router(top2 gather lists) -> per-expert bf16 WMMA GEMM (double-
// buffered LDS, XOR-swizzled, packed bf16 conversion) + GELU + gated scatter.
// ---------------------------------------------------------------------------

typedef __bf16 bf16_t;
typedef __attribute__((ext_vector_type(16))) __bf16 v16bf;
typedef __attribute__((ext_vector_type(8)))  __bf16 v8bf;
typedef __attribute__((ext_vector_type(8)))  float  v8f;

constexpr int H_  = 2048;
constexpr int E_  = 8;

constexpr int BM = 128;   // tokens per block tile
constexpr int BN = 128;   // output features per block tile
constexpr int BK = 32;    // k-depth per stage (one bf16 WMMA K)
constexpr int WM = 64;    // per-wave M tile (2 waves in M)
constexpr int WN = 32;    // per-wave N tile (4 waves in N)

__device__ __forceinline__ bf16_t to_bf16(float f) { return (bf16_t)f; }

// XOR-swizzled element offset inside a [rows][BK] bf16 tile.
// 16B granule index (k>>3) is XORed with (row>>2)&3 so that the per-lane
// 16B ds_load_b128 fragment reads of a 16-row tile hit 16 distinct bank
// groups, while granule contents stay contiguous (b128-aligned).
__device__ __forceinline__ int swz(int row, int k) {
  return row * BK + (((k & 24) ^ ((row & 12) << 1)) | (k & 7));
}

// ---------------------------------------------------------------------------
// Kernel 0: zero output + expert counters
// ---------------------------------------------------------------------------
__global__ void moe_zero_kernel(float4* __restrict__ out4, long n4,
                                int* __restrict__ counts) {
  long i = (long)blockIdx.x * blockDim.x + threadIdx.x;
  long stride = (long)gridDim.x * blockDim.x;
  float4 z; z.x = 0.f; z.y = 0.f; z.z = 0.f; z.w = 0.f;
  for (long j = i; j < n4; j += stride) out4[j] = z;
  if (i < 16) counts[i] = 0;
}

// ---------------------------------------------------------------------------
// Kernel 1: router. One wave (32 lanes) per token.
// ---------------------------------------------------------------------------
__global__ __launch_bounds__(256)
void moe_router_kernel(const float* __restrict__ x,
                       const float* __restrict__ Wr,
                       const float* __restrict__ br,
                       int T,
                       int* __restrict__ counts,
                       int* __restrict__ tokids,
                       float* __restrict__ gates) {
  const int lane = threadIdx.x & 31;
  const int wave = threadIdx.x >> 5;
  const int t = blockIdx.x * (blockDim.x >> 5) + wave;
  if (t >= T) return;

  float acc[E_];
#pragma unroll
  for (int e = 0; e < E_; ++e) acc[e] = 0.f;

  const float* xr = x + (size_t)t * H_;
  for (int h = lane; h < H_; h += 32) {
    float xv = xr[h];
    const float4* w4 = (const float4*)(Wr + (size_t)h * E_);
    float4 w0 = w4[0], w1 = w4[1];
    acc[0] += xv * w0.x; acc[1] += xv * w0.y;
    acc[2] += xv * w0.z; acc[3] += xv * w0.w;
    acc[4] += xv * w1.x; acc[5] += xv * w1.y;
    acc[6] += xv * w1.z; acc[7] += xv * w1.w;
  }
#pragma unroll
  for (int off = 16; off >= 1; off >>= 1) {
#pragma unroll
    for (int e = 0; e < E_; ++e) acc[e] += __shfl_xor(acc[e], off, 32);
  }

  if (lane == 0) {
    float m = -1e30f;
#pragma unroll
    for (int e = 0; e < E_; ++e) { acc[e] += br[e]; m = fmaxf(m, acc[e]); }
    float p[E_];
    float s = 0.f;
#pragma unroll
    for (int e = 0; e < E_; ++e) { p[e] = expf(acc[e] - m); s += p[e]; }
    float inv_s = 1.f / s;
#pragma unroll
    for (int e = 0; e < E_; ++e) p[e] *= inv_s;

    int i1 = 0;
#pragma unroll
    for (int e = 1; e < E_; ++e) if (p[e] > p[i1]) i1 = e;
    int i2 = (i1 == 0) ? 1 : 0;
#pragma unroll
    for (int e = 0; e < E_; ++e) if (e != i1 && p[e] > p[i2]) i2 = e;

    float v1 = p[i1], v2 = p[i2];
    float inv = 1.f / (v1 + v2);

    int pos1 = atomicAdd(&counts[i1], 1);
    tokids[i1 * T + pos1] = t;
    gates [i1 * T + pos1] = v1 * inv;
    int pos2 = atomicAdd(&counts[i2], 1);
    tokids[i2 * T + pos2] = t;
    gates [i2 * T + pos2] = v2 * inv;
  }
}

// ---------------------------------------------------------------------------
// Kernel 2: per-expert gathered GEMM, bf16 WMMA, double-buffered LDS.
// grid = (H/BN, ceil(T/BM), E), block = 256 (8 waves: 2 in M x 4 in N)
// ---------------------------------------------------------------------------
__global__ __launch_bounds__(256)
void moe_gemm_kernel(const float* __restrict__ x,
                     const float* __restrict__ We,
                     const float* __restrict__ be,
                     const int* __restrict__ counts,
                     const int* __restrict__ tokids,
                     const float* __restrict__ gates,
                     float* __restrict__ out,
                     int T) {
  const int e   = blockIdx.z;
  const int cnt = counts[e];
  const int m0  = blockIdx.y * BM;
  if (m0 >= cnt) return;
  const int n0  = blockIdx.x * BN;

  __shared__ bf16_t sA[2][BM * BK];   // [buf][swizzled row*BK + k]
  __shared__ bf16_t sB[2][BN * BK];   // [buf][swizzled n*BK + k] (B transposed)
  __shared__ int    sTok[BM];
  __shared__ float  sGate[BM];

  const int tid  = threadIdx.x;
  const int lane = tid & 31;
  const int wave = tid >> 5;
  const int wm   = wave & 1;          // 0..1
  const int wn   = wave >> 1;         // 0..3

  if (tid < BM) {
    int r = m0 + tid;
    if (r < cnt) { sTok[tid] = tokids[e * T + r]; sGate[tid] = gates[e * T + r]; }
    else         { sTok[tid] = -1;                sGate[tid] = 0.f; }
  }
  __syncthreads();

  const float* Wbase = We + (size_t)e * H_ * H_;

  v8f acc[4][2];
#pragma unroll
  for (int mi = 0; mi < 4; ++mi)
#pragma unroll
    for (int ni = 0; ni < 2; ++ni)
      acc[mi][ni] = (v8f){0.f, 0.f, 0.f, 0.f, 0.f, 0.f, 0.f, 0.f};

  // Fragment addressing (ISA 16-bit A/B layout, wave32):
  // lanes 0-15: K in {0..7}U{16..23}; lanes 16-31: K in {8..15}U{24..31}
  const int fr   = lane & 15;
  const int koff = (lane >> 4) * 8;

  // A staging map: 2 threads per token row, 16 consecutive k each
  const int a_row = tid >> 1;
  const int a_kk  = (tid & 1) * 16;
  const int a_tok = sTok[a_row];
  // B staging map: thread handles 2 k-pairs x 4 consecutive n
  const int b_kp  = tid >> 5;         // 0..7 (k-pair group; +8 for 2nd unit)
  const int b_n   = (lane) * 4;       // 0..124

  float4 pa[4];                       // A prefetch regs (16 floats)
  float4 pb0[2], pb1[2];              // B prefetch regs (2 units x 2 rows)

  // ---- global -> regs for one K-stage ----
  auto load_regs = [&](int k0) {
    if (a_tok >= 0) {
      const float4* src = (const float4*)(x + (size_t)a_tok * H_ + k0 + a_kk);
#pragma unroll
      for (int j = 0; j < 4; ++j) pa[j] = src[j];
    } else {
#pragma unroll
      for (int j = 0; j < 4; ++j) { pa[j].x = pa[j].y = pa[j].z = pa[j].w = 0.f; }
    }
#pragma unroll
    for (int u = 0; u < 2; ++u) {
      int k = (b_kp + u * 8) * 2;
      const float* s = Wbase + (size_t)(k0 + k) * H_ + n0 + b_n;
      pb0[u] = *(const float4*)s;
      pb1[u] = *(const float4*)(s + H_);
    }
  };

  // ---- regs -> LDS (packed bf16, swizzled) ----
  auto store_lds = [&](int b) {
    const float* f = (const float*)pa;
    v8bf v0, v1;
#pragma unroll
    for (int j = 0; j < 8; ++j) { v0[j] = to_bf16(f[j]); v1[j] = to_bf16(f[8 + j]); }
    *(v8bf*)&sA[b][swz(a_row, a_kk)]     = v0;
    *(v8bf*)&sA[b][swz(a_row, a_kk + 8)] = v1;
#pragma unroll
    for (int u = 0; u < 2; ++u) {
      int k = (b_kp + u * 8) * 2;
      const float* f0 = (const float*)&pb0[u];
      const float* f1 = (const float*)&pb1[u];
#pragma unroll
      for (int c = 0; c < 4; ++c) {
        union { bf16_t h[2]; unsigned u32; } pk;
        pk.h[0] = to_bf16(f0[c]);   // K = k
        pk.h[1] = to_bf16(f1[c]);   // K = k+1
        *(unsigned*)&sB[b][swz(b_n + c, k)] = pk.u32;
      }
    }
  };

  // ---- LDS -> fragments -> 8x WMMA ----
  auto compute = [&](int b) {
    v16bf afrag[4], bfrag[2];
#pragma unroll
    for (int mi = 0; mi < 4; ++mi) {
      int row = wm * WM + mi * 16 + fr;
      union { v16bf v; v8bf h[2]; } u;
      u.h[0] = *(const v8bf*)&sA[b][swz(row, koff)];
      u.h[1] = *(const v8bf*)&sA[b][swz(row, 16 + koff)];
      afrag[mi] = u.v;
    }
#pragma unroll
    for (int ni = 0; ni < 2; ++ni) {
      int col = wn * WN + ni * 16 + fr;
      union { v16bf v; v8bf h[2]; } u;
      u.h[0] = *(const v8bf*)&sB[b][swz(col, koff)];
      u.h[1] = *(const v8bf*)&sB[b][swz(col, 16 + koff)];
      bfrag[ni] = u.v;
    }
#pragma unroll
    for (int mi = 0; mi < 4; ++mi)
#pragma unroll
      for (int ni = 0; ni < 2; ++ni)
        acc[mi][ni] = __builtin_amdgcn_wmma_f32_16x16x32_bf16(
            false, afrag[mi], false, bfrag[ni],
            (short)0, acc[mi][ni], false, false);
  };

  // ---- double-buffered main loop: one barrier per K-stage ----
  load_regs(0);
  store_lds(0);
  int buf = 0;
  for (int k0 = 0; k0 < H_; k0 += BK) {
    __syncthreads();                      // lds[buf] produced by all waves
    const bool has_next = (k0 + BK) < H_;
    if (has_next) load_regs(k0 + BK);     // global loads issue before WMMAs
    compute(buf);                         // overlap HBM/L2 latency
    if (has_next) store_lds(buf ^ 1);     // other buffer: no extra barrier
    buf ^= 1;
  }

  // ---- epilogue: bias + exact GELU + gate, scatter with f32 atomics ----
  // C/D layout: lane 0-15 -> N=lane, M=v ; lane 16-31 -> N=lane-16, M=8+v
  const int nlane = lane & 15;
  const int mhalf = (lane >> 4) * 8;
#pragma unroll
  for (int mi = 0; mi < 4; ++mi) {
    int rbase = wm * WM + mi * 16 + mhalf;
    int   toks[8];
    float gts[8];
#pragma unroll
    for (int v = 0; v < 8; ++v) { toks[v] = sTok[rbase + v]; gts[v] = sGate[rbase + v]; }
#pragma unroll
    for (int ni = 0; ni < 2; ++ni) {
      int col  = n0 + wn * WN + ni * 16 + nlane;
      float bias = be[e * H_ + col];
#pragma unroll
      for (int v = 0; v < 8; ++v) {
        if (toks[v] >= 0) {
          float val  = acc[mi][ni][v] + bias;
          float gelu = 0.5f * val * (1.f + erff(val * 0.70710678118654752f));
          atomicAdd(out + (size_t)toks[v] * H_ + col, gts[v] * gelu);
        }
      }
    }
  }
}

// ---------------------------------------------------------------------------
// Host launcher
// ---------------------------------------------------------------------------
extern "C" void kernel_launch(void* const* d_in, const int* in_sizes, int n_in,
                              void* d_out, int out_size, void* d_ws, size_t ws_size,
                              hipStream_t stream) {
  const float* x  = (const float*)d_in[0];
  const float* We = (const float*)d_in[1];
  const float* be = (const float*)d_in[2];
  const float* Wr = (const float*)d_in[3];
  const float* br = (const float*)d_in[4];
  (void)n_in; (void)ws_size; (void)out_size;

  const int T = in_sizes[0] / H_;            // B*S tokens
  float* out = (float*)d_out;

  // workspace layout: [16 ints counts][E*T ints tokids][E*T floats gates]
  int*   counts = (int*)d_ws;
  int*   tokids = counts + 16;
  float* gates  = (float*)(tokids + (size_t)E_ * T);

  long n4 = (long)T * H_ / 4;
  moe_zero_kernel<<<2048, 256, 0, stream>>>((float4*)out, n4, counts);

  int waves_per_blk = 256 / 32;
  moe_router_kernel<<<(T + waves_per_blk - 1) / waves_per_blk, 256, 0, stream>>>(
      x, Wr, br, T, counts, tokids, gates);

  dim3 grid(H_ / BN, (T + BM - 1) / BM, E_);
  moe_gemm_kernel<<<grid, 256, 0, stream>>>(
      x, We, be, counts, tokids, gates, out, T);
}
